// VQVAE_87265145520458
// MI455X (gfx1250) — compile-verified, matrix operand order checked by reading, not run on hardware
//
#include <hip/hip_runtime.h>
#include <hip/hip_bf16.h>

// ---------------------------------------------------------------------------
// VQ-VAE forward for MI455X (gfx1250, wave32, WMMA bf16 16x16x32 + TDM).
//   * argmin_k(||z||^2 + ||e_k||^2) is independent of z (faithful reference
//     bug) -> z_q is one 128-vector broadcast everywhere.
//   * decoder input is batch/space constant -> decoder runs on ONE image,
//     result broadcast to all 32 outputs.
//   * z_e never materialized: vq loss fused into the e3 WMMA kernel.
// Heavy convs (e2, e3, dt1) are implicit GEMMs on v_wmma_f32_16x16x32_bf16,
// A (weight) panels DMA'd to LDS with tensor_load_to_lds each K-step.
// ---------------------------------------------------------------------------

typedef __attribute__((ext_vector_type(16))) __bf16          v16bf;
typedef __attribute__((ext_vector_type(8)))  float           v8f;
typedef __attribute__((ext_vector_type(8)))  unsigned short  ushort8;
typedef __attribute__((ext_vector_type(4)))  unsigned int    u32x4;
typedef __attribute__((ext_vector_type(8)))  int             i32x8;
typedef __attribute__((ext_vector_type(4)))  int             i32x4;

#if defined(__has_builtin)
#  if __has_builtin(__builtin_amdgcn_tensor_load_to_lds) && __has_builtin(__builtin_amdgcn_s_wait_tensorcnt)
#    define HAVE_TDM 1
#  else
#    define HAVE_TDM 0
#  endif
#else
#  define HAVE_TDM 0
#endif

__device__ __forceinline__ unsigned short f2bf(float f) {
    union { float f; unsigned u; } v; v.f = f;
    unsigned r = v.u + 0x7FFFu + ((v.u >> 16) & 1u);   // RNE
    return (unsigned short)(r >> 16);
}
__device__ __forceinline__ float bf2f(unsigned short h) {
    union { unsigned u; float f; } v; v.u = ((unsigned)h) << 16;
    return v.f;
}

// K index held by (lane, element e) of the 16-bit A/B fragment (ISA 7.12.2).
// Simplifies to: K = (e<8) ? half*8+e : 16+half*8+(e-8), half = lane>>4.
__device__ __forceinline__ int kmap(int l, int e) {
    int base = (l >> 4) * 8;
    return (e < 8) ? (base + e) : (16 + base + (e - 8));
}

// 2D tile DMA: rows x row_elems u16, row stride in elems, into LDS offset.
__device__ __forceinline__ void tdm_load_2d(unsigned lds_off, const void* gaddr,
                                            unsigned row_elems, unsigned rows,
                                            unsigned row_stride_elems) {
#if HAVE_TDM
    unsigned long long ga = (unsigned long long)gaddr;
    u32x4 g0;
    g0[0] = 1u;                                            // count=1 (valid D#)
    g0[1] = lds_off;                                       // lds_addr (bytes)
    g0[2] = (unsigned)ga;                                  // global_addr[31:0]
    g0[3] = (unsigned)(ga >> 32) | 0x80000000u;            // addr[56:32] | type=2
    i32x8 g1;
    g1[0] = 1 << 16;                                       // data_size=1 (2B)
    g1[1] = (int)((row_elems & 0xFFFFu) << 16);            // tensor_dim0 lo16
    g1[2] = (int)((row_elems >> 16) | ((rows & 0xFFFFu) << 16)); // dim0 hi | dim1 lo
    g1[3] = (int)((rows >> 16) | (row_elems << 16));       // dim1 hi | tile_dim0
    g1[4] = (int)rows;                                     // tile_dim1 (tile_dim2=0)
    g1[5] = (int)row_stride_elems;                         // tensor_dim0_stride lo32
    g1[6] = 0;
    g1[7] = 0;
    i32x4 z4 = {0, 0, 0, 0};
#  if __clang_major__ >= 23
    i32x8 z8 = {0, 0, 0, 0, 0, 0, 0, 0};
    __builtin_amdgcn_tensor_load_to_lds(g0, g1, z4, z4, z8, 0);
#  else
    __builtin_amdgcn_tensor_load_to_lds(g0, g1, z4, z4, 0);
#  endif
#else
    (void)lds_off; (void)gaddr; (void)row_elems; (void)rows; (void)row_stride_elems;
#endif
}

// ---------------------------------------------------------------------------
// Weight pre-pack: A fragments so each lane's 16 bf16 are contiguous:
//   A[((mt*KT + kt)*32 + lane)*16 + e]
// ---------------------------------------------------------------------------
__global__ void k_pack_e2(const float* __restrict__ w, unsigned short* __restrict__ A) {
    int t = blockIdx.x * 256 + threadIdx.x;                 // 8*32*512
    if (t >= 8 * 32 * 512) return;
    int e = t & 15, l = (t >> 4) & 31, kt = (t >> 9) & 31, mt = t >> 14;
    int k = kt * 32 + kmap(l, e);
    int ci = k & 63, tap = k >> 6, ky = tap >> 2, kx = tap & 3;
    int co = mt * 16 + (l & 15);
    A[t] = f2bf(w[((co * 64 + ci) * 4 + ky) * 4 + kx]);     // w_e2 [128,64,4,4]
}

__global__ void k_pack_e3(const float* __restrict__ w, unsigned short* __restrict__ A) {
    int t = blockIdx.x * 256 + threadIdx.x;                 // 8*36*512
    if (t >= 8 * 36 * 512) return;
    int e = t & 15, l = (t >> 4) & 31;
    int rest = t >> 9;
    int kt = rest % 36, mt = rest / 36;
    int k = kt * 32 + kmap(l, e);
    int ci = k & 127, tap = k >> 7, ky = tap / 3, kx = tap % 3;
    int co = mt * 16 + (l & 15);
    A[t] = f2bf(w[((co * 128 + ci) * 3 + ky) * 3 + kx]);    // w_e3 [128,128,3,3]
}

__global__ void k_pack_dt1(const float* __restrict__ w, unsigned short* __restrict__ A) {
    int t = blockIdx.x * 256 + threadIdx.x;                 // 4*64*512
    if (t >= 4 * 64 * 512) return;
    int e = t & 15, l = (t >> 4) & 31, kt = (t >> 9) & 63, mt = t >> 15;
    int k = kt * 32 + kmap(l, e);
    int ci = k & 127, tap = k >> 7, ky = tap >> 2, kx = tap & 3;
    int co = mt * 16 + (l & 15);
    A[t] = f2bf(w[((ci * 64 + co) * 4 + ky) * 4 + kx]);     // w_dt1 [128,64,4,4] (Cin,Cout)
}

// ---------------------------------------------------------------------------
// Codebook argmin (first index on ties, like jnp.argmin) + selected row.
// ---------------------------------------------------------------------------
__global__ void k_codebook(const float* __restrict__ emb,
                           float* __restrict__ zq_f,
                           unsigned short* __restrict__ zq_bf) {
    __shared__ float nrm[512];
    __shared__ int   idx[512];
    int t = threadIdx.x;
    float s = 0.f;
    const float* row = emb + t * 128;
    for (int c = 0; c < 128; ++c) { float v = row[c]; s += v * v; }
    nrm[t] = s; idx[t] = t;
    __syncthreads();
    for (int off = 256; off; off >>= 1) {
        if (t < off) {
            float a = nrm[t], b = nrm[t + off];
            if (b < a || (b == a && idx[t + off] < idx[t])) { nrm[t] = b; idx[t] = idx[t + off]; }
        }
        __syncthreads();
    }
    int best = idx[0];
    if (t < 128) { float v = emb[best * 128 + t]; zq_f[t] = v; zq_bf[t] = f2bf(v); }
}

// wsum[co][tap] = sum_ci w_d1[co,ci,ky,kx] * zq[ci]
__global__ void k_wsum(const float* __restrict__ w_d1, const float* __restrict__ zq,
                       float* __restrict__ wsum) {
    int t = blockIdx.x * 256 + threadIdx.x;
    if (t >= 128 * 9) return;
    int co = t / 9, tap = t % 9, ky = tap / 3, kx = tap % 3;
    float s = 0.f;
    for (int ci = 0; ci < 128; ++ci)
        s += w_d1[((co * 128 + ci) * 3 + ky) * 3 + kx] * zq[ci];
    wsum[t] = s;
}

// ---------------------------------------------------------------------------
// e1: conv 1->64, 4x4 s2 p1, relu -> bf16 h1 [32,64,112,112]
// ---------------------------------------------------------------------------
__global__ void k_e1(const float* __restrict__ x, const float* __restrict__ w,
                     const float* __restrict__ b, unsigned short* __restrict__ h1) {
    __shared__ float wl[1024];
    __shared__ float bl[64];
    int tid = threadIdx.x;
    for (int i = tid; i < 1024; i += 256) wl[i] = w[i];
    if (tid < 64) bl[tid] = b[tid];
    __syncthreads();
    int t = blockIdx.x * 256 + tid;
    if (t >= 32 * 112 * 112) return;
    int img = t / 12544, sp = t % 12544, oy = sp / 112, ox = sp % 112;
    float xv[16];
    #pragma unroll
    for (int ky = 0; ky < 4; ++ky)
        #pragma unroll
        for (int kx = 0; kx < 4; ++kx) {
            int iy = oy * 2 - 1 + ky, ix = ox * 2 - 1 + kx;
            xv[ky * 4 + kx] = ((unsigned)iy < 224u && (unsigned)ix < 224u)
                            ? x[(img * 224 + iy) * 224 + ix] : 0.f;
        }
    for (int co = 0; co < 64; ++co) {
        float s = bl[co];
        #pragma unroll
        for (int k = 0; k < 16; ++k) s += wl[co * 16 + k] * xv[k];
        s = s > 0.f ? s : 0.f;
        h1[((img * 64 + co) * 112 + oy) * 112 + ox] = f2bf(s);
    }
}

// ---------------------------------------------------------------------------
// e2: conv 64->128 4x4 s2 p1 relu. GEMM M=128 N=100352 K=1024.
// Block: 8 waves (one M-tile each) x 4 N-subtiles (N=64/block).
// A panel (8x1024B) DMA'd to LDS via TDM each K-step; B im2col gathered as
// 8-consecutive-channel strided loads + one ds_store_b128 per thread.
// ---------------------------------------------------------------------------
__global__ void k_e2(const unsigned short* __restrict__ h1,
                     const unsigned short* __restrict__ Apack,
                     const float* __restrict__ bias,
                     unsigned short* __restrict__ h2) {
    __shared__ __align__(32) unsigned short ldsA[8 * 512];
    __shared__ __align__(32) unsigned short ldsB[4 * 512];
    const int tid = threadIdx.x, wave = tid >> 5, lane = tid & 31;
    const int nbase = blockIdx.x * 64;
    const int KT = 32;
    // gather mapping: thread owns (subtile st, column col, K-octet q)
    const int cg = tid & 63, st = cg >> 4, col = cg & 15, q = tid >> 6;
    const int l = (q & 1) * 16 + col, ebase = (q & 2) ? 8 : 0;
    const int n = nbase + st * 16 + col;
    const int img = n / 3136, sp = n % 3136;
    const int oy = sp / 56, ox = sp % 56;
    const int iy0 = oy * 2 - 1, ix0 = ox * 2 - 1;
    const unsigned short* inb = h1 + (size_t)img * 64 * 12544;
    unsigned short* ldsw = ldsB + (st * 512 + l * 16 + ebase);

    v8f cacc[4];
    #pragma unroll
    for (int i = 0; i < 4; ++i) cacc[i] = (v8f){0.f,0.f,0.f,0.f,0.f,0.f,0.f,0.f};

    for (int kt = 0; kt < KT; ++kt) {
#if HAVE_TDM
        if (tid < 32)       // one TDM issue per block (EXEC-independent op)
            tdm_load_2d((unsigned)(unsigned long long)(void*)ldsA,
                        Apack + (size_t)kt * 512, 512, 8, KT * 512);
#endif
        {   // B gather: ci = (kt&1)*32 + q*8 + j  (8 consecutive channels)
            int tap = kt >> 1, ky = tap >> 2, kx = tap & 3;
            int ci0 = (kt & 1) * 32 + q * 8;
            int iy = iy0 + ky, ix = ix0 + kx;
            ushort8 v = (ushort8){0,0,0,0,0,0,0,0};
            if ((unsigned)iy < 112u && (unsigned)ix < 112u) {
                const unsigned short* s = inb + (size_t)ci0 * 12544 + iy * 112 + ix;
                #pragma unroll
                for (int j = 0; j < 8; ++j) v[j] = s[(size_t)j * 12544];
            }
            *(ushort8*)ldsw = v;
        }
#if HAVE_TDM
        if (tid < 32) __builtin_amdgcn_s_wait_tensorcnt(0);
#endif
        __syncthreads();
#if HAVE_TDM
        v16bf a = *(const v16bf*)(ldsA + wave * 512 + lane * 16);
#else
        v16bf a = *(const v16bf*)(Apack + ((size_t)(wave * KT + kt) * 32 + lane) * 16);
#endif
        #pragma unroll
        for (int s2 = 0; s2 < 4; ++s2) {
            v16bf bfr = *(const v16bf*)(ldsB + s2 * 512 + lane * 16);
            cacc[s2] = __builtin_amdgcn_wmma_f32_16x16x32_bf16(false, a, false, bfr,
                                                               (short)0, cacc[s2], false, false);
        }
        __syncthreads();
    }
    const int halfl = lane >> 4, coln = lane & 15;
    #pragma unroll
    for (int s2 = 0; s2 < 4; ++s2) {
        int sp2 = (nbase + s2 * 16 + coln) % 3136;   // img constant per block
        #pragma unroll
        for (int r = 0; r < 8; ++r) {
            int co = wave * 16 + r + 8 * halfl;
            float v = cacc[s2][r] + bias[co];
            v = v > 0.f ? v : 0.f;
            h2[(size_t)(img * 128 + co) * 3136 + sp2] = f2bf(v);
        }
    }
}

// ---------------------------------------------------------------------------
// e3: conv 128->128 3x3 s1 p1 (z_e) with fused vq loss. K=1152.
// ---------------------------------------------------------------------------
__global__ void k_e3_loss(const unsigned short* __restrict__ h2,
                          const unsigned short* __restrict__ Apack,
                          const float* __restrict__ bias,
                          const float* __restrict__ zq,
                          float* __restrict__ vq_acc) {
    __shared__ __align__(32) unsigned short ldsA[8 * 512];
    __shared__ __align__(32) unsigned short ldsB[4 * 512];
    __shared__ float red[256];
    const int tid = threadIdx.x, wave = tid >> 5, lane = tid & 31;
    const int nbase = blockIdx.x * 64;
    const int KT = 36;
    const int cg = tid & 63, st = cg >> 4, col = cg & 15, q = tid >> 6;
    const int l = (q & 1) * 16 + col, ebase = (q & 2) ? 8 : 0;
    const int n = nbase + st * 16 + col;
    const int img = n / 3136, sp = n % 3136;
    const int oy = sp / 56, ox = sp % 56;
    const unsigned short* inb = h2 + (size_t)img * 128 * 3136;
    unsigned short* ldsw = ldsB + (st * 512 + l * 16 + ebase);

    v8f cacc[4];
    #pragma unroll
    for (int i = 0; i < 4; ++i) cacc[i] = (v8f){0.f,0.f,0.f,0.f,0.f,0.f,0.f,0.f};

    for (int kt = 0; kt < KT; ++kt) {
#if HAVE_TDM
        if (tid < 32)
            tdm_load_2d((unsigned)(unsigned long long)(void*)ldsA,
                        Apack + (size_t)kt * 512, 512, 8, KT * 512);
#endif
        {   // ci = (kt&3)*32 + q*8 + j ; tap = kt>>2
            int tap = kt >> 2, ky = tap / 3, kx = tap % 3;
            int ci0 = (kt & 3) * 32 + q * 8;
            int iy = oy - 1 + ky, ix = ox - 1 + kx;
            ushort8 v = (ushort8){0,0,0,0,0,0,0,0};
            if ((unsigned)iy < 56u && (unsigned)ix < 56u) {
                const unsigned short* s = inb + (size_t)ci0 * 3136 + iy * 56 + ix;
                #pragma unroll
                for (int j = 0; j < 8; ++j) v[j] = s[(size_t)j * 3136];
            }
            *(ushort8*)ldsw = v;
        }
#if HAVE_TDM
        if (tid < 32) __builtin_amdgcn_s_wait_tensorcnt(0);
#endif
        __syncthreads();
#if HAVE_TDM
        v16bf a = *(const v16bf*)(ldsA + wave * 512 + lane * 16);
#else
        v16bf a = *(const v16bf*)(Apack + ((size_t)(wave * KT + kt) * 32 + lane) * 16);
#endif
        #pragma unroll
        for (int s2 = 0; s2 < 4; ++s2) {
            v16bf bfr = *(const v16bf*)(ldsB + s2 * 512 + lane * 16);
            cacc[s2] = __builtin_amdgcn_wmma_f32_16x16x32_bf16(false, a, false, bfr,
                                                               (short)0, cacc[s2], false, false);
        }
        __syncthreads();
    }
    const int halfl = lane >> 4;
    float acc = 0.f;
    #pragma unroll
    for (int s2 = 0; s2 < 4; ++s2) {
        #pragma unroll
        for (int r = 0; r < 8; ++r) {
            int co = wave * 16 + r + 8 * halfl;
            float z = cacc[s2][r] + bias[co];
            float d = zq[co] - z;
            acc += d * d;
        }
    }
    red[tid] = acc;
    __syncthreads();
    for (int off = 128; off; off >>= 1) {
        if (tid < off) red[tid] += red[tid + off];
        __syncthreads();
    }
    if (tid == 0) atomicAdd(vq_acc, red[0]);
}

// d1 on constant z_q: only 3x3 border-validity structure matters. One image.
__global__ void k_d1(const float* __restrict__ wsum, const float* __restrict__ b_d1,
                     unsigned short* __restrict__ d_img) {
    int t = blockIdx.x * 256 + threadIdx.x;
    if (t >= 128 * 3136) return;
    int co = t / 3136, sp = t % 3136, oy = sp / 56, ox = sp % 56;
    float s = b_d1[co];
    #pragma unroll
    for (int ky = 0; ky < 3; ++ky) {
        int iy = oy - 1 + ky; if ((unsigned)iy >= 56u) continue;
        #pragma unroll
        for (int kx = 0; kx < 3; ++kx) {
            int ix = ox - 1 + kx; if ((unsigned)ix >= 56u) continue;
            s += wsum[co * 9 + ky * 3 + kx];
        }
    }
    d_img[t] = f2bf(s);
}

// ---------------------------------------------------------------------------
// dt1: ConvTranspose 128->64 4x4 s2 p1 relu, ONE image. GEMM M=64 N=12544
// K=2048 (invalid taps zero-filled). 4 waves, N=64/block.
// ---------------------------------------------------------------------------
__global__ void k_dt1(const unsigned short* __restrict__ d_img,
                      const unsigned short* __restrict__ Apack,
                      const float* __restrict__ bias,
                      unsigned short* __restrict__ t1) {
    __shared__ __align__(32) unsigned short ldsA[4 * 512];
    __shared__ __align__(32) unsigned short ldsB[4 * 512];
    const int tid = threadIdx.x, wave = tid >> 5, lane = tid & 31;  // 128 thr
    const int nbase = blockIdx.x * 64;
    const int KT = 64;
    const int cg = tid & 63, st = cg >> 4, col = cg & 15, qh = tid >> 6; // qh 0..1
    const int n = nbase + st * 16 + col;
    const int oy = n / 112, ox = n % 112;

    v8f cacc[4];
    #pragma unroll
    for (int i = 0; i < 4; ++i) cacc[i] = (v8f){0.f,0.f,0.f,0.f,0.f,0.f,0.f,0.f};

    for (int kt = 0; kt < KT; ++kt) {
#if HAVE_TDM
        if (tid < 32)
            tdm_load_2d((unsigned)(unsigned long long)(void*)ldsA,
                        Apack + (size_t)kt * 512, 512, 4, KT * 512);
#endif
        {
            int tap = kt >> 2, ky = tap >> 2, kx = tap & 3;
            int ty = oy + 1 - ky, tx = ox + 1 - kx;
            bool valid = (ty >= 0) && !(ty & 1) && (tx >= 0) && !(tx & 1)
                       && ((ty >> 1) < 56) && ((tx >> 1) < 56);
            int iy = ty >> 1, ix = tx >> 1;
            #pragma unroll
            for (int h = 0; h < 2; ++h) {          // each thread covers 2 K-octets
                int q = qh + 2 * h;
                int l = (q & 1) * 16 + col, ebase = (q & 2) ? 8 : 0;
                int ci0 = (kt & 3) * 32 + q * 8;
                ushort8 v = (ushort8){0,0,0,0,0,0,0,0};
                if (valid) {
                    const unsigned short* s = d_img + (size_t)ci0 * 3136 + iy * 56 + ix;
                    #pragma unroll
                    for (int j = 0; j < 8; ++j) v[j] = s[(size_t)j * 3136];
                }
                *(ushort8*)(ldsB + st * 512 + l * 16 + ebase) = v;
            }
        }
#if HAVE_TDM
        if (tid < 32) __builtin_amdgcn_s_wait_tensorcnt(0);
#endif
        __syncthreads();
#if HAVE_TDM
        v16bf a = *(const v16bf*)(ldsA + wave * 512 + lane * 16);
#else
        v16bf a = *(const v16bf*)(Apack + ((size_t)(wave * KT + kt) * 32 + lane) * 16);
#endif
        #pragma unroll
        for (int s2 = 0; s2 < 4; ++s2) {
            v16bf bfr = *(const v16bf*)(ldsB + s2 * 512 + lane * 16);
            cacc[s2] = __builtin_amdgcn_wmma_f32_16x16x32_bf16(false, a, false, bfr,
                                                               (short)0, cacc[s2], false, false);
        }
        __syncthreads();
    }
    const int halfl = lane >> 4, coln = lane & 15;
    #pragma unroll
    for (int s2 = 0; s2 < 4; ++s2) {
        int n2 = nbase + s2 * 16 + coln;
        #pragma unroll
        for (int r = 0; r < 8; ++r) {
            int co = wave * 16 + r + 8 * halfl;
            float v = cacc[s2][r] + bias[co];
            v = v > 0.f ? v : 0.f;
            t1[co * 12544 + n2] = f2bf(v);
        }
    }
}

// dt2: ConvTranspose 64->1 4x4 s2 p1, one image -> rec [224,224] f32
__global__ void k_dt2(const unsigned short* __restrict__ t1, const float* __restrict__ w,
                      const float* __restrict__ b, float* __restrict__ rec) {
    int t = blockIdx.x * 256 + threadIdx.x;
    if (t >= 224 * 224) return;
    int oy = t / 224, ox = t % 224;
    float s = b[0];
    for (int ci = 0; ci < 64; ++ci) {
        #pragma unroll
        for (int ky = 0; ky < 4; ++ky) {
            int ty = oy + 1 - ky;
            if (ty < 0 || (ty & 1)) continue;
            int iy = ty >> 1; if (iy >= 112) continue;
            #pragma unroll
            for (int kx = 0; kx < 4; ++kx) {
                int tx = ox + 1 - kx;
                if (tx < 0 || (tx & 1)) continue;
                int ix = tx >> 1; if (ix >= 112) continue;
                s += bf2f(t1[(ci * 112 + iy) * 112 + ix]) * w[ci * 16 + ky * 4 + kx];
            }
        }
    }
    rec[t] = s;
}

// broadcast rec to all 32 outputs + fused reconstruction-loss reduction
__global__ void k_out(const float* __restrict__ rec, const float* __restrict__ x,
                      float* __restrict__ out, float* __restrict__ rec_acc) {
    __shared__ float red[256];
    int t = blockIdx.x * 256 + threadIdx.x;
    float a = 0.f;
    if (t < 32 * 50176) {
        int p = t % 50176;
        float r = rec[p];
        out[t] = r;
        float d = r - x[t];
        a = d * d;
    }
    red[threadIdx.x] = a;
    __syncthreads();
    for (int off = 128; off; off >>= 1) {
        if (threadIdx.x < off) red[threadIdx.x] += red[threadIdx.x + off];
        __syncthreads();
    }
    if (threadIdx.x == 0) atomicAdd(rec_acc, red[0]);
}

__global__ void k_zero(float* acc) {
    if (threadIdx.x < 2) acc[threadIdx.x] = 0.f;
}

__global__ void k_final(const float* __restrict__ acc, float* __restrict__ out_tail) {
    float vq = 2.f * acc[0] / (float)(32 * 128 * 56 * 56);
    float rc = acc[1] / (float)(32 * 224 * 224);
    out_tail[0] = rc + vq;
    out_tail[1] = rc;
    out_tail[2] = vq;
}

// ---------------------------------------------------------------------------
extern "C" void kernel_launch(void* const* d_in, const int* in_sizes, int n_in,
                              void* d_out, int out_size, void* d_ws, size_t ws_size,
                              hipStream_t stream) {
    const float* x     = (const float*)d_in[0];
    const float* w_e1  = (const float*)d_in[1];
    const float* b_e1  = (const float*)d_in[2];
    const float* w_e2  = (const float*)d_in[3];
    const float* b_e2  = (const float*)d_in[4];
    const float* w_e3  = (const float*)d_in[5];
    const float* b_e3  = (const float*)d_in[6];
    const float* emb   = (const float*)d_in[7];
    const float* w_d1  = (const float*)d_in[8];
    const float* b_d1  = (const float*)d_in[9];
    const float* w_dt1 = (const float*)d_in[10];
    const float* b_dt1 = (const float*)d_in[11];
    const float* w_dt2 = (const float*)d_in[12];
    const float* b_dt2 = (const float*)d_in[13];
    float* out = (float*)d_out;

    char* ws = (char*)d_ws;
    size_t off = 0;
    auto carve = [&](size_t bytes) -> char* {
        char* p = ws + off;
        off = (off + bytes + 255) & ~(size_t)255;
        return p;
    };
    float*          acc   = (float*)carve(8);
    float*          zq_f  = (float*)carve(128 * 4);
    unsigned short* zq_bf = (unsigned short*)carve(128 * 2);
    float*          wsum  = (float*)carve(1152 * 4);
    float*          rec   = (float*)carve(50176 * 4);
    unsigned short* Ae2   = (unsigned short*)carve((size_t)8 * 32 * 512 * 2);
    unsigned short* Ae3   = (unsigned short*)carve((size_t)8 * 36 * 512 * 2);
    unsigned short* Adt1  = (unsigned short*)carve((size_t)4 * 64 * 512 * 2);
    unsigned short* h1    = (unsigned short*)carve((size_t)32 * 64 * 112 * 112 * 2);
    unsigned short* h2    = (unsigned short*)carve((size_t)32 * 128 * 56 * 56 * 2);
    unsigned short* d_img = (unsigned short*)carve((size_t)128 * 56 * 56 * 2);
    unsigned short* t1    = (unsigned short*)carve((size_t)64 * 112 * 112 * 2);
    (void)ws_size; (void)in_sizes; (void)n_in; (void)out_size; (void)zq_bf;

    k_zero<<<1, 32, 0, stream>>>(acc);
    k_pack_e2 <<<512, 256, 0, stream>>>(w_e2,  Ae2);
    k_pack_e3 <<<576, 256, 0, stream>>>(w_e3,  Ae3);
    k_pack_dt1<<<512, 256, 0, stream>>>(w_dt1, Adt1);
    k_codebook<<<1, 512, 0, stream>>>(emb, zq_f, zq_bf);
    k_wsum<<<5, 256, 0, stream>>>(w_d1, zq_f, wsum);

    k_e1<<<1568, 256, 0, stream>>>(x, w_e1, b_e1, h1);
    k_e2<<<1568, 256, 0, stream>>>(h1, Ae2, b_e2, h2);                  // WMMA+TDM
    k_e3_loss<<<1568, 256, 0, stream>>>(h2, Ae3, b_e3, zq_f, &acc[0]);  // WMMA+TDM

    k_d1<<<1568, 256, 0, stream>>>(wsum, b_d1, d_img);
    k_dt1<<<196, 128, 0, stream>>>(d_img, Adt1, b_dt1, t1);             // WMMA+TDM
    k_dt2<<<196, 256, 0, stream>>>(t1, w_dt2, b_dt2, rec);

    k_out<<<6272, 256, 0, stream>>>(rec, x, out, &acc[1]);
    k_final<<<1, 1, 0, stream>>>(acc, out + 32 * 50176);
}
